// MILClassifier_64312840290706
// MI455X (gfx1250) — matrix operand order repaired and checked by reading, not compiled, not run on hardware
//
#include <hip/hip_runtime.h>
#include <cstdint>
#include <cstddef>

// ---------------------------------------------------------------------------
// Types for CDNA5 WMMA (wave32, 16x16x32 bf16 -> f32)
// ---------------------------------------------------------------------------
typedef __attribute__((ext_vector_type(16))) __bf16 v16bf_t;
typedef __attribute__((ext_vector_type(8)))  __bf16 v8bf_t;
typedef __attribute__((ext_vector_type(4)))  __bf16 v4bf_t;
typedef __attribute__((ext_vector_type(8)))  float  v8f_t;

#define N_INST 262144
#define N_BAGS 16
#define D_IN   1024
#define H1DIM  512
#define H2DIM  256
#define H3DIM  128
#define D_ATT  128
#define M_TILE 64
#define THREADS 256
#define CPB    16   // pooling chunks per bag

// LDS layout (bytes). Regions are reused phase-to-phase (with barriers):
//   phase1: Xs [0,131072)            + H1s [131072,196608)  (written)
//   phase2: H1s read, H2s [0,32768)  (written; Xs dead)
//   phase3: H2s read, H3s [32768,49152)
//   phase4: H3s read, scorePart [49152,51200) written
#define LDS_XS    0
#define LDS_H1    131072
#define LDS_H2    0
#define LDS_H3    32768
#define LDS_SCORE 49152
#define LDS_BYTES 196608

__device__ __forceinline__ v8f_t wmma_bf16(v16bf_t a, v16bf_t b, v8f_t c) {
  // 8 args: (neg_a, A, neg_b, B, c_mod, C, reuse_a, reuse_b)
  return __builtin_amdgcn_wmma_f32_16x16x32_bf16(false, a, false, b, (short)0, c,
                                                 false, false);
}

// Load an A fragment (16x32 bf16, row-major in LDS, row stride ldx elements).
// Per ISA A-layout: lanes 0-15 hold M=0..15 with K 0..7 (V0-3) / 16..23 (V4-7),
// lanes 16-31 hold K 8..15 / 24..31.  -> two contiguous 16B reads per lane.
__device__ __forceinline__ v16bf_t load_a_frag(const __bf16* __restrict__ Xlds,
                                               int ldx, int row, int kbase) {
  const __bf16* p = Xlds + row * ldx + kbase;
  v8bf_t lo = *(const v8bf_t*)p;
  v8bf_t hi = *(const v8bf_t*)(p + 16);
  return __builtin_shufflevector(lo, hi, 0, 1, 2, 3, 4, 5, 6, 7,
                                 8, 9, 10, 11, 12, 13, 14, 15);
}

// ---------------------------------------------------------------------------
// Generic fused layer over a 64-row tile (4 row tiles of 16).
// Each wave owns coltiles nt = wave + 8*i (i < NTPW); coltiles are processed
// in groups of NP (4 when possible, else 2, else 1) so each A fragment feeds
// NP WMMAs -- this divides LDS A-traffic by NP and is what balances LDS
// bandwidth (256B/clk/WGP) against the 4 matrix pipes.
// MODE: 0 = ReLU->bf16 LDS, 2 = + global h3 store,
// 3 = attention epilogue (tanh * Wv[col], shuffle-reduce, per-wave partial).
// ---------------------------------------------------------------------------
template <int NKT, int NTPW, int MODE>
__device__ __forceinline__ void layer(const __bf16* __restrict__ Xlds, int ldx,
                                      const __bf16* __restrict__ Wp,
                                      const float* __restrict__ bias,
                                      __bf16* __restrict__ Olds, int ldo,
                                      __bf16* __restrict__ h3g,
                                      float* scorePart,
                                      const float* __restrict__ Wv,
                                      int wave, int lane) {
  constexpr int NP = (NTPW % 4 == 0) ? 4 : (NTPW % 2 == 0) ? 2 : 1;
  const int g  = lane >> 4;
  const int ln = lane & 15;
#pragma unroll
  for (int i = 0; i < NTPW; i += NP) {
    v8f_t acc[NP][4] = {};
    for (int kt = 0; kt < NKT; ++kt) {
      v16bf_t bfrag[NP];
#pragma unroll
      for (int p = 0; p < NP; ++p) {
        const int nt = wave + 8 * (i + p);
        bfrag[p] = *(const v16bf_t*)(Wp + (((size_t)(nt * NKT + kt)) << 9) +
                                     (lane << 4));
      }
      const int kb = kt * 32 + 8 * g;
#pragma unroll
      for (int rt = 0; rt < 4; ++rt) {
        v16bf_t afrag = load_a_frag(Xlds, ldx, rt * 16 + ln, kb);
#pragma unroll
        for (int p = 0; p < NP; ++p)
          acc[p][rt] = wmma_bf16(afrag, bfrag[p], acc[p][rt]);
      }
    }
#pragma unroll
    for (int p = 0; p < NP; ++p) {
      const int nt = wave + 8 * (i + p);
      const float bb = bias[nt * 16 + ln];
#pragma unroll
      for (int rt = 0; rt < 4; ++rt) {
#pragma unroll
        for (int v = 0; v < 8; ++v) {
          float val = acc[p][rt][v] + bb;
          const int row = rt * 16 + v + 8 * g;  // C layout: M = v + 8*(lane/16)
          const int col = nt * 16 + ln;         // N = lane%16
          if constexpr (MODE == 3) {
            val = tanhf(val) * Wv[col];
            // deterministic reduce over the 16 lanes sharing this row
            val += __shfl_xor(val, 1);
            val += __shfl_xor(val, 2);
            val += __shfl_xor(val, 4);
            val += __shfl_xor(val, 8);
            if (ln == 0) scorePart[wave * M_TILE + row] = val;
          } else {
            val = fmaxf(val, 0.0f);
            Olds[row * ldo + col] = (__bf16)val;
            if constexpr (MODE == 2) {
              h3g[(size_t)row * H3DIM + col] = (__bf16)val;
            }
          }
        }
      }
    }
  }
}

// ---------------------------------------------------------------------------
// Weight packer: fp32 (K x N, row-major) -> bf16 WMMA B-fragment layout.
// Tile (nt,kt) is 512 bf16; lane L holds 16 consecutive elements.
// B layout: lane L -> N-in-tile = L%16; element s -> K-in-tile = 8*(L/16) + (s&7) + 16*(s>>3).
// ---------------------------------------------------------------------------
__global__ __launch_bounds__(256) void pack_weights_kernel(
    const float* __restrict__ W, int K, int N, __bf16* __restrict__ out) {
  const int nkt = K >> 5;
  const int total = K * N;
  const int idx = blockIdx.x * blockDim.x + threadIdx.x;
  if (idx >= total) return;
  const int s = idx & 15;
  const int L = (idx >> 4) & 31;
  const int t = idx >> 9;
  const int kt = t % nkt;
  const int nt = t / nkt;
  const int k = kt * 32 + 8 * (L >> 4) + (s & 7) + ((s >> 3) << 4);
  const int n = nt * 16 + (L & 15);
  out[idx] = (__bf16)W[(size_t)k * N + n];
}

// ---------------------------------------------------------------------------
// Main fused kernel: one block = 64 instances, 8 wave32.
// ---------------------------------------------------------------------------
__global__ __launch_bounds__(THREADS, 1) void mil_fused_kernel(
    const float* __restrict__ x,
    const __bf16* __restrict__ W1p, const float* __restrict__ b1,
    const __bf16* __restrict__ W2p, const float* __restrict__ b2,
    const __bf16* __restrict__ W3p, const float* __restrict__ b3,
    const __bf16* __restrict__ Wap, const float* __restrict__ ba,
    const float* __restrict__ Wv, const float* __restrict__ bvv,
    __bf16* __restrict__ h3g, float* __restrict__ scores) {
  extern __shared__ char smem[];
  __bf16* Xs  = (__bf16*)(smem + LDS_XS);
  __bf16* H1s = (__bf16*)(smem + LDS_H1);
  __bf16* H2s = (__bf16*)(smem + LDS_H2);
  __bf16* H3s = (__bf16*)(smem + LDS_H3);
  float* scorePart = (float*)(smem + LDS_SCORE);  // [8 waves][64 rows]

  const int tid = threadIdx.x;
  const int wave = tid >> 5;
  const int lane = tid & 31;
  const int inst0 = blockIdx.x * M_TILE;

  // Stage x tile (64 x 1024 fp32) -> LDS bf16, float4 loads.
  const float4* x4 = (const float4*)(x + (size_t)inst0 * D_IN);
  for (int t = tid; t < M_TILE * (D_IN / 4); t += THREADS) {
    const int row = t >> 8;  // D_IN/4 == 256
    const int c4 = t & 255;
    float4 v = x4[(size_t)row * (D_IN / 4) + c4];
    v4bf_t pk = {(__bf16)v.x, (__bf16)v.y, (__bf16)v.z, (__bf16)v.w};
    *(v4bf_t*)(Xs + row * D_IN + c4 * 4) = pk;
  }
  __syncthreads();

  // h1 = relu(x@W1 + b1) : K=1024 (32 ktiles), N=512 (coltile QUAD per wave)
  layer<32, 4, 0>(Xs, D_IN, W1p, b1, H1s, H1DIM, nullptr, nullptr, nullptr,
                  wave, lane);
  __syncthreads();
  // h2 = relu(h1@W2 + b2) : K=512, N=256 (1 coltile-pair/wave)
  layer<16, 2, 0>(H1s, H1DIM, W2p, b2, H2s, H2DIM, nullptr, nullptr, nullptr,
                  wave, lane);
  __syncthreads();
  // h3 = relu(h2@W3 + b3) : K=256, N=128 ; also spilled to HBM for pooling
  layer<8, 1, 2>(H2s, H2DIM, W3p, b3, H3s, H3DIM,
                 h3g + (size_t)inst0 * H3DIM, nullptr, nullptr, wave, lane);
  __syncthreads();
  // score partials: tanh(h3@Wa + ba) * Wv, per-wave column-strip sums
  layer<4, 1, 3>(H3s, H3DIM, Wap, ba, nullptr, 0, nullptr, scorePart, Wv, wave,
                 lane);
  __syncthreads();
  if (tid < M_TILE) {
    float s = bvv[0];
#pragma unroll
    for (int w = 0; w < 8; ++w) s += scorePart[w * M_TILE + tid];
    scores[inst0 + tid] = s;
  }
}

// ---------------------------------------------------------------------------
// Per-bag softmax: rewrite scores -> normalized attention weights in place.
// ---------------------------------------------------------------------------
__global__ __launch_bounds__(256) void bag_softmax_kernel(
    float* __restrict__ scores, const long long* __restrict__ bag_sizes) {
  __shared__ float red[256];
  const int b = blockIdx.x;
  const int tid = threadIdx.x;
  long long start = 0;
  for (int j = 0; j < b; ++j) start += bag_sizes[j];
  const long long size = bag_sizes[b];

  float m = -3.402823466e38f;
  for (long long i = tid; i < size; i += 256) m = fmaxf(m, scores[start + i]);
  red[tid] = m;
  __syncthreads();
  for (int s = 128; s > 0; s >>= 1) {
    if (tid < s) red[tid] = fmaxf(red[tid], red[tid + s]);
    __syncthreads();
  }
  m = red[0];
  __syncthreads();

  float sum = 0.0f;
  for (long long i = tid; i < size; i += 256) sum += expf(scores[start + i] - m);
  red[tid] = sum;
  __syncthreads();
  for (int s = 128; s > 0; s >>= 1) {
    if (tid < s) red[tid] += red[tid + s];
    __syncthreads();
  }
  const float denom = red[0];

  for (long long i = tid; i < size; i += 256)
    scores[start + i] = expf(scores[start + i] - m) / denom;
}

// ---------------------------------------------------------------------------
// Chunked weighted pooling: block (b,c) accumulates its slice of bag b and
// writes a deterministic per-chunk partial feature vector (no atomics).
// ---------------------------------------------------------------------------
__global__ __launch_bounds__(256) void bag_accum_kernel(
    const __bf16* __restrict__ h3, const float* __restrict__ w,
    const long long* __restrict__ bag_sizes, float* __restrict__ partial) {
  __shared__ float red[256];
  const int b = blockIdx.x / CPB;
  const int c = blockIdx.x % CPB;
  long long start = 0;
  for (int j = 0; j < b; ++j) start += bag_sizes[j];
  const long long size = bag_sizes[b];
  const long long i0 = (size * c) / CPB;
  const long long i1 = (size * (c + 1)) / CPB;

  const int t = threadIdx.x;
  const int d = t & 127;
  const int sub = t >> 7;
  float acc = 0.0f;
  for (long long i = i0 + sub; i < i1; i += 2)
    acc += w[start + i] * (float)h3[(size_t)(start + i) * H3DIM + d];
  red[t] = acc;
  __syncthreads();
  if (sub == 0)
    partial[((size_t)b * CPB + c) * H3DIM + d] = red[t] + red[t + 128];
}

// ---------------------------------------------------------------------------
// Final: out[b] = (sum_c partial[b][c]) @ Wc + bc   (deterministic)
// ---------------------------------------------------------------------------
__global__ __launch_bounds__(128) void final_out_kernel(
    const float* __restrict__ partial, const float* __restrict__ Wc,
    const float* __restrict__ bc, float* __restrict__ out) {
  __shared__ float red[128];
  const int b = blockIdx.x;
  const int d = threadIdx.x;
  float s = 0.0f;
#pragma unroll
  for (int c = 0; c < CPB; ++c)
    s += partial[((size_t)b * CPB + c) * H3DIM + d];
  red[d] = s * Wc[d];
  __syncthreads();
  for (int st = 64; st > 0; st >>= 1) {
    if (d < st) red[d] += red[d + st];
    __syncthreads();
  }
  if (d == 0) out[b] = red[0] + bc[0];
}

// ---------------------------------------------------------------------------
extern "C" void kernel_launch(void* const* d_in, const int* in_sizes, int n_in,
                              void* d_out, int out_size, void* d_ws,
                              size_t ws_size, hipStream_t stream) {
  (void)in_sizes; (void)n_in; (void)out_size; (void)ws_size;
  const float* x = (const float*)d_in[0];
  const long long* bag_sizes = (const long long*)d_in[1];
  const float* W1 = (const float*)d_in[2];
  const float* b1 = (const float*)d_in[3];
  const float* W2 = (const float*)d_in[4];
  const float* b2 = (const float*)d_in[5];
  const float* W3 = (const float*)d_in[6];
  const float* b3 = (const float*)d_in[7];
  const float* Wa = (const float*)d_in[8];
  const float* ba = (const float*)d_in[9];
  const float* Wv = (const float*)d_in[10];
  const float* bv = (const float*)d_in[11];
  const float* Wc = (const float*)d_in[12];
  const float* bc = (const float*)d_in[13];
  float* out = (float*)d_out;

  char* ws = (char*)d_ws;
  size_t off = 0;
  __bf16* h3g = (__bf16*)(ws + off);    off += (size_t)N_INST * H3DIM * 2;  // 64 MB
  float* scores = (float*)(ws + off);   off += (size_t)N_INST * 4;          // 1 MB
  __bf16* W1p = (__bf16*)(ws + off);    off += (size_t)D_IN * H1DIM * 2;
  __bf16* W2p = (__bf16*)(ws + off);    off += (size_t)H1DIM * H2DIM * 2;
  __bf16* W3p = (__bf16*)(ws + off);    off += (size_t)H2DIM * H3DIM * 2;
  __bf16* Wap = (__bf16*)(ws + off);    off += (size_t)H3DIM * D_ATT * 2;
  float* partial = (float*)(ws + off);  off += (size_t)N_BAGS * CPB * H3DIM * 4;

  pack_weights_kernel<<<(D_IN * H1DIM + 255) / 256, 256, 0, stream>>>(W1, D_IN, H1DIM, W1p);
  pack_weights_kernel<<<(H1DIM * H2DIM + 255) / 256, 256, 0, stream>>>(W2, H1DIM, H2DIM, W2p);
  pack_weights_kernel<<<(H2DIM * H3DIM + 255) / 256, 256, 0, stream>>>(W3, H2DIM, H3DIM, W3p);
  pack_weights_kernel<<<(H3DIM * D_ATT + 255) / 256, 256, 0, stream>>>(Wa, H3DIM, D_ATT, Wap);

  (void)hipFuncSetAttribute((const void*)mil_fused_kernel,
                            hipFuncAttributeMaxDynamicSharedMemorySize,
                            LDS_BYTES);

  mil_fused_kernel<<<N_INST / M_TILE, THREADS, LDS_BYTES, stream>>>(
      x, W1p, b1, W2p, b2, W3p, b3, Wap, ba, Wv, bv, h3g, scores);

  bag_softmax_kernel<<<N_BAGS, 256, 0, stream>>>(scores, bag_sizes);
  bag_accum_kernel<<<N_BAGS * CPB, 256, 0, stream>>>(h3g, scores, bag_sizes, partial);
  final_out_kernel<<<N_BAGS, 128, 0, stream>>>(partial, Wc, bc, out);
}
